// RfillAutoreg_80083960201302
// MI455X (gfx1250) — matrix operand order repaired and checked by reading, not compiled, not run on hardware
//
#include <hip/hip_runtime.h>

typedef __bf16 bf16;
typedef __attribute__((ext_vector_type(8)))  __bf16 v8bf;
typedef __attribute__((ext_vector_type(16))) __bf16 v16bf;
typedef __attribute__((ext_vector_type(8)))  float  v8f;

#define E_DIM 128
#define V_DIM 128
#define T_LEN 128
#define B_SZ  8192
#define BM    32      // batch rows per block
#define AP    136     // padded LDS pitch (bf16 elems) for A-operand staging
#define LP    132     // padded logits pitch (floats)
#define NTHREADS 256  // 8 waves

__device__ __forceinline__ float sigf(float x){ return 1.0f/(1.0f + __expf(-x)); }

// A-fragment (16x32 bf16) from row-major LDS tile with pitch AP.
// Lane L: row = L%16, K = kc*32 + (L/16)*8 + {0..7} and +16..+23.
__device__ __forceinline__ v16bf ldsA(const bf16* base, int kc){
  int lane = threadIdx.x & 31;
  int row  = lane & 15;
  int kb   = kc*32 + ((lane>>4)<<3);
  const bf16* p = base + row*AP + kb;
  union { v16bf v; struct { v8bf lo; v8bf hi; } s; } u;
  u.s.lo = *(const v8bf*)p;
  u.s.hi = *(const v8bf*)(p + 16);
  return u.v;
}

// B-fragment (32x16 bf16) from pre-swizzled LDS weights.
// Fragment (kc,nt) stored as [chunk(2)][lane(32)][8 elems].
__device__ __forceinline__ v16bf ldsB(const bf16* wlds, int kc, int nt){
  int lane = threadIdx.x & 31;
  const bf16* f = wlds + (kc*32 + nt)*512;
  union { v16bf v; struct { v8bf lo; v8bf hi; } s; } u;
  u.s.lo = *(const v8bf*)(f + lane*8);
  u.s.hi = *(const v8bf*)(f + 256 + lane*8);
  return u.v;
}

__device__ __forceinline__ v8f wmma_bf16(v16bf a, v16bf b, v8f c){
  return __builtin_amdgcn_wmma_f32_16x16x32_bf16(false, a, false, b, (short)0, c, false, false);
}

// Convert f32 global weights into bf16 B-fragment layout in LDS.
// Combined K: kc 0..3 -> Wih (K=0..127), kc 4..7 -> Whh (K=0..127).
__device__ __forceinline__ void fill_w(bf16* wlds, const float* Wih, const float* Whh, int nkc){
  int ntask = nkc*32*32; // frags * lanes
  for (int task = threadIdx.x; task < ntask; task += NTHREADS){
    int ln   = task & 31;
    int frag = task >> 5;
    int nt   = frag & 31;
    int kc   = frag >> 5;
    int n    = nt*16 + (ln & 15);
    int kb   = (kc & 3)*32 + ((ln>>4)<<3);
    const float* src = ((kc < 4) ? Wih : Whh) + n*E_DIM + kb;  // W[n][k], k contiguous
    bf16* dst = wlds + frag*512;
    #pragma unroll
    for (int i=0;i<8;i++) dst[ln*8 + i]       = (bf16)src[i];
    #pragma unroll
    for (int i=0;i<8;i++) dst[256 + ln*8 + i] = (bf16)src[16+i];
  }
}

// ---------------- Kernel 1: one-step LSTM from h=c=0 (init state) ----------------
template<bool IN_BF16>
__global__ __launch_bounds__(NTHREADS)
void lstm_init_step(const void* xin, const float* Wih,
                    const float* bih, const float* bhh,
                    bf16* Hout, float* Cout)
{
  __shared__ bf16 wlds[4*32*512];   // 128 KB: Wih fragments
  __shared__ bf16 xA[BM*AP];
  int tid = threadIdx.x, lane = tid & 31, wave = tid >> 5;
  int row0 = blockIdx.x * BM;

  fill_w(wlds, Wih, Wih, 4);
  for (int idx = tid; idx < BM*E_DIM; idx += NTHREADS){
    int r = idx >> 7, c = idx & 127;
    float v;
    if (IN_BF16) v = (float)((const bf16*)xin)[(size_t)(row0+r)*E_DIM + c];
    else         v = ((const float*)xin)[(size_t)(row0+r)*E_DIM + c];
    xA[r*AP + c] = (bf16)v;
  }
  __syncthreads();

  float bias[4];
  #pragma unroll
  for (int j=0;j<4;j++){
    int n = (wave + 8*j)*16 + (lane & 15);
    bias[j] = bih[n] + bhh[n];
  }

  v8f acc[2][4];
  #pragma unroll
  for (int s=0;s<2;s++)
    #pragma unroll
    for (int j=0;j<4;j++)
      #pragma unroll
      for (int r=0;r<8;r++) acc[s][j][r] = bias[j];

  #pragma unroll
  for (int kc=0;kc<4;kc++){
    v16bf a0 = ldsA(xA, kc);
    v16bf a1 = ldsA(xA + 16*AP, kc);
    #pragma unroll
    for (int j=0;j<4;j++){
      v16bf b = ldsB(wlds, kc, wave + 8*j);
      acc[0][j] = wmma_bf16(a0, b, acc[0][j]);
      acc[1][j] = wmma_bf16(a1, b, acc[1][j]);
    }
  }

  int e = wave*16 + (lane & 15);
  #pragma unroll
  for (int s=0;s<2;s++){
    #pragma unroll
    for (int r=0;r<8;r++){
      int row = s*16 + r + ((lane>>4)<<3);
      float gi = acc[s][0][r], gg = acc[s][2][r], go = acc[s][3][r];
      float cs = sigf(gi) * tanhf(gg);   // c0 = 0 (f-gate term vanishes)
      float h  = sigf(go) * tanhf(cs);
      size_t o = (size_t)(row0+row)*E_DIM + e;
      Cout[o] = cs;
      Hout[o] = (bf16)h;
    }
  }
}

// ---------------- Kernel 2: layer-0 recurrence over all T (embed gather in LDS) ----------------
__global__ __launch_bounds__(NTHREADS)
void lstm_layer0_seq(const int* tokens, const float* embed,
                     const float* Wih, const float* Whh,
                     const float* bih, const float* bhh,
                     const bf16* Hinit, const float* Cinit,
                     bf16* Hseq)
{
  __shared__ bf16 wlds[8*32*512];     // 256 KB: [Wih;Whh] fragments
  __shared__ bf16 elds[V_DIM*AP];     // embedding table (bf16)
  __shared__ bf16 hA[BM*AP];          // h_t staging (A-operand)
  __shared__ int  tk[BM];
  int tid = threadIdx.x, lane = tid & 31, wave = tid >> 5;
  int row0 = blockIdx.x * BM;

  fill_w(wlds, Wih, Whh, 8);
  for (int idx = tid; idx < V_DIM*E_DIM; idx += NTHREADS){
    int r = idx >> 7, c = idx & 127;
    elds[r*AP + c] = (bf16)embed[idx];
  }
  for (int idx = tid; idx < BM*E_DIM; idx += NTHREADS){
    int r = idx >> 7, c = idx & 127;
    hA[r*AP + c] = Hinit[(size_t)(row0+r)*E_DIM + c];
  }

  float bias[4];
  #pragma unroll
  for (int j=0;j<4;j++){
    int n = (wave + 8*j)*16 + (lane & 15);
    bias[j] = bih[n] + bhh[n];
  }
  int e = wave*16 + (lane & 15);
  float cst[16];
  #pragma unroll
  for (int s=0;s<2;s++)
    #pragma unroll
    for (int r=0;r<8;r++)
      cst[s*8+r] = Cinit[(size_t)(row0 + s*16 + r + ((lane>>4)<<3))*E_DIM + e];

  for (int t=0; t<T_LEN; ++t){
    if (tid < BM) tk[tid] = tokens[(size_t)(row0+tid)*T_LEN + t];
    __syncthreads();

    v8f acc[2][4];
    #pragma unroll
    for (int s=0;s<2;s++)
      #pragma unroll
      for (int j=0;j<4;j++)
        #pragma unroll
        for (int r=0;r<8;r++) acc[s][j][r] = bias[j];

    #pragma unroll
    for (int kc=0;kc<8;kc++){
      v16bf a0, a1;
      if (kc < 4){
        int rr = lane & 15;
        int kb = kc*32 + ((lane>>4)<<3);
        union { v16bf v; struct { v8bf lo; v8bf hi; } s; } u;
        const bf16* p0 = elds + tk[rr]*AP + kb;
        u.s.lo = *(const v8bf*)p0; u.s.hi = *(const v8bf*)(p0 + 16);
        a0 = u.v;
        const bf16* p1 = elds + tk[16+rr]*AP + kb;
        u.s.lo = *(const v8bf*)p1; u.s.hi = *(const v8bf*)(p1 + 16);
        a1 = u.v;
      } else {
        a0 = ldsA(hA, kc-4);
        a1 = ldsA(hA + 16*AP, kc-4);
      }
      #pragma unroll
      for (int j=0;j<4;j++){
        v16bf b = ldsB(wlds, kc, wave + 8*j);
        acc[0][j] = wmma_bf16(a0, b, acc[0][j]);
        acc[1][j] = wmma_bf16(a1, b, acc[1][j]);
      }
    }
    __syncthreads();   // all hA reads done before overwrite

    #pragma unroll
    for (int s=0;s<2;s++){
      #pragma unroll
      for (int r=0;r<8;r++){
        int row = s*16 + r + ((lane>>4)<<3);
        float gi = acc[s][0][r], gf = acc[s][1][r], gg = acc[s][2][r], go = acc[s][3][r];
        float cs = sigf(gf)*cst[s*8+r] + sigf(gi)*tanhf(gg);
        float h  = sigf(go)*tanhf(cs);
        cst[s*8+r] = cs;
        hA[row*AP + e] = (bf16)h;
      }
    }
    __syncthreads();   // hA complete

    { // coalesced h_t -> global
      int r = tid >> 3, c0 = (tid & 7) * 16;
      v8bf x0 = *(const v8bf*)(hA + r*AP + c0);
      v8bf x1 = *(const v8bf*)(hA + r*AP + c0 + 8);
      bf16* dst = Hseq + (size_t)t*B_SZ*E_DIM + (size_t)(row0+r)*E_DIM + c0;
      *(v8bf*)dst = x0;
      *(v8bf*)(dst + 8) = x1;
    }
  }
}

// ---------------- Kernel 3: layer-1 recurrence + fused logits/log-softmax/NLL ----------------
__global__ __launch_bounds__(NTHREADS)
void lstm_layer1_loss(const bf16* Hseq, const float* Wih, const float* Whh,
                      const float* bih, const float* bhh,
                      const bf16* Hinit, const float* Cinit,
                      const float* Wout, const float* bout,
                      const int* tokens, float* ll)
{
  __shared__ bf16 wlds[8*32*512];     // 256 KB
  __shared__ bf16 xA[BM*AP];
  __shared__ bf16 hA[BM*AP];
  __shared__ float lbuf[BM*LP];
  __shared__ float red1[BM*8];
  __shared__ float red2[BM*8];
  int tid = threadIdx.x, lane = tid & 31, wave = tid >> 5;
  int row0 = blockIdx.x * BM;

  fill_w(wlds, Wih, Whh, 8);
  for (int idx = tid; idx < BM*E_DIM; idx += NTHREADS){
    int r = idx >> 7, c = idx & 127;
    hA[r*AP + c] = Hinit[(size_t)(row0+r)*E_DIM + c];
  }

  float bias[4];
  #pragma unroll
  for (int j=0;j<4;j++){
    int n = (wave + 8*j)*16 + (lane & 15);
    bias[j] = bih[n] + bhh[n];
  }
  int e = wave*16 + (lane & 15);
  float cst[16];
  #pragma unroll
  for (int s=0;s<2;s++)
    #pragma unroll
    for (int r=0;r<8;r++)
      cst[s*8+r] = Cinit[(size_t)(row0 + s*16 + r + ((lane>>4)<<3))*E_DIM + e];

  // W_out B-fragments held in registers (ntile == wave), 4 kchunks over E=128
  v16bf wf[4];
  #pragma unroll
  for (int kc=0;kc<4;kc++){
    int n  = wave*16 + (lane & 15);
    int kb = kc*32 + ((lane>>4)<<3);
    const float* src = Wout + n*E_DIM + kb;
    v16bf w;
    #pragma unroll
    for (int i=0;i<16;i++) w[i] = (bf16)src[(i<8) ? i : i+8];
    wf[kc] = w;
  }
  float bo = bout[wave*16 + (lane & 15)];

  int Trow = tid >> 3, Tj = tid & 7;
  float llacc = 0.0f;

  for (int t=0; t<T_LEN-1; ++t){
    { // stage x_t = Hseq[t] tile
      int r = tid >> 3, c0 = (tid & 7) * 16;
      const bf16* src = Hseq + (size_t)t*B_SZ*E_DIM + (size_t)(row0+r)*E_DIM + c0;
      v8bf x0 = *(const v8bf*)src;
      v8bf x1 = *(const v8bf*)(src + 8);
      *(v8bf*)(xA + r*AP + c0) = x0;
      *(v8bf*)(xA + r*AP + c0 + 8) = x1;
    }
    __syncthreads();

    v8f acc[2][4];
    #pragma unroll
    for (int s=0;s<2;s++)
      #pragma unroll
      for (int j=0;j<4;j++)
        #pragma unroll
        for (int r=0;r<8;r++) acc[s][j][r] = bias[j];

    #pragma unroll
    for (int kc=0;kc<8;kc++){
      v16bf a0 = (kc < 4) ? ldsA(xA, kc)         : ldsA(hA, kc-4);
      v16bf a1 = (kc < 4) ? ldsA(xA + 16*AP, kc) : ldsA(hA + 16*AP, kc-4);
      #pragma unroll
      for (int j=0;j<4;j++){
        v16bf b = ldsB(wlds, kc, wave + 8*j);
        acc[0][j] = wmma_bf16(a0, b, acc[0][j]);
        acc[1][j] = wmma_bf16(a1, b, acc[1][j]);
      }
    }
    __syncthreads();

    #pragma unroll
    for (int s=0;s<2;s++){
      #pragma unroll
      for (int r=0;r<8;r++){
        int row = s*16 + r + ((lane>>4)<<3);
        float gi = acc[s][0][r], gf = acc[s][1][r], gg = acc[s][2][r], go = acc[s][3][r];
        float cs = sigf(gf)*cst[s*8+r] + sigf(gi)*tanhf(gg);
        float h  = sigf(go)*tanhf(cs);
        cst[s*8+r] = cs;
        hA[row*AP + e] = (bf16)h;
      }
    }
    __syncthreads();

    // logits = h @ Wout^T + b_out  (A from hA, B from registers)
    v8f lacc[2];
    #pragma unroll
    for (int s=0;s<2;s++)
      #pragma unroll
      for (int r=0;r<8;r++) lacc[s][r] = bo;
    #pragma unroll
    for (int kc=0;kc<4;kc++){
      v16bf a0 = ldsA(hA, kc);
      v16bf a1 = ldsA(hA + 16*AP, kc);
      lacc[0] = wmma_bf16(a0, wf[kc], lacc[0]);
      lacc[1] = wmma_bf16(a1, wf[kc], lacc[1]);
    }
    int v = wave*16 + (lane & 15);
    #pragma unroll
    for (int s=0;s<2;s++)
      #pragma unroll
      for (int r=0;r<8;r++)
        lbuf[(s*16 + r + ((lane>>4)<<3))*LP + v] = lacc[s][r];
    __syncthreads();

    // log-softmax + NLL: 8 threads per row
    float mx = -1e30f;
    #pragma unroll
    for (int i=0;i<16;i++) mx = fmaxf(mx, lbuf[Trow*LP + Tj*16 + i]);
    red1[Trow*8 + Tj] = mx;
    __syncthreads();
    float m = red1[Trow*8];
    #pragma unroll
    for (int i=1;i<8;i++) m = fmaxf(m, red1[Trow*8 + i]);
    float se = 0.0f;
    #pragma unroll
    for (int i=0;i<16;i++) se += __expf(lbuf[Trow*LP + Tj*16 + i] - m);
    red2[Trow*8 + Tj] = se;
    __syncthreads();
    if (Tj == 0){
      float ssum = 0.0f;
      #pragma unroll
      for (int i=0;i<8;i++) ssum += red2[Trow*8 + i];
      int tgt = tokens[(size_t)(row0+Trow)*T_LEN + (t+1)];
      float lt = lbuf[Trow*LP + tgt];
      float nll = (m + __logf(ssum)) - lt;
      if (tgt != 0) llacc -= nll;   // ignore_index = PAD(0)
    }
  }

  if (Tj == 0) ll[row0 + Trow] = llacc;
}

extern "C" void kernel_launch(void* const* d_in, const int* in_sizes, int n_in,
                              void* d_out, int out_size, void* d_ws, size_t ws_size,
                              hipStream_t stream)
{
  const float* state  = (const float*)d_in[0];
  const int*   tokens = (const int*)  d_in[1];
  const float* embed  = (const float*)d_in[2];
  const float* Wih    = (const float*)d_in[3];
  const float* Whh    = (const float*)d_in[4];
  const float* bih    = (const float*)d_in[5];
  const float* bhh    = (const float*)d_in[6];
  const float* Wout   = (const float*)d_in[7];
  const float* bout   = (const float*)d_in[8];
  float* out = (float*)d_out;

  char* ws = (char*)d_ws;
  bf16*  Hinit0 = (bf16*) ws; ws += (size_t)B_SZ*E_DIM*sizeof(bf16);
  float* Cinit0 = (float*)ws; ws += (size_t)B_SZ*E_DIM*sizeof(float);
  bf16*  Hinit1 = (bf16*) ws; ws += (size_t)B_SZ*E_DIM*sizeof(bf16);
  float* Cinit1 = (float*)ws; ws += (size_t)B_SZ*E_DIM*sizeof(float);
  bf16*  Hseq0  = (bf16*) ws; ws += (size_t)T_LEN*B_SZ*E_DIM*sizeof(bf16);

  dim3 grid(B_SZ/BM), block(NTHREADS);
  const int WL = 4*E_DIM*E_DIM;  // per-layer weight stride (floats)
  const int BL = 4*E_DIM;        // per-layer bias stride

  lstm_init_step<false><<<grid, block, 0, stream>>>(state,  Wih,      bih,      bhh,      Hinit0, Cinit0);
  lstm_init_step<true ><<<grid, block, 0, stream>>>(Hinit0, Wih + WL, bih + BL, bhh + BL, Hinit1, Cinit1);
  lstm_layer0_seq<<<grid, block, 0, stream>>>(tokens, embed, Wih, Whh, bih, bhh, Hinit0, Cinit0, Hseq0);
  lstm_layer1_loss<<<grid, block, 0, stream>>>(Hseq0, Wih + WL, Whh + WL, bih + BL, bhh + BL,
                                               Hinit1, Cinit1, Wout, bout, tokens, out);
}
